// BayesianSG_34892314313727
// MI455X (gfx1250) — matrix operand order repaired and checked by reading, not compiled, not run on hardware
//
#include <hip/hip_runtime.h>
#include <stdint.h>

// MI455X / gfx1250, wave32, WMMA (not MFMA).
typedef __attribute__((ext_vector_type(16))) _Float16 v16h;
typedef __attribute__((ext_vector_type(8)))  float    v8f;

#define DEVFN __device__ __forceinline__

namespace {

constexpr int kV = 50000;   // vocab
constexpr int kD = 128;     // embed dim
constexpr int kB = 2048;    // batch
constexpr int kC = 20;      // context window
constexpr int kE = 256;     // 2*D encoder width
constexpr int kTiles = kV / 16;   // 3125 column tiles of 16
constexpr int kCsplit = 4;        // column groups per 16-row tile (occupancy)
constexpr int kWaves  = 8;        // waves per block in the WMMA kernel
constexpr int kStride = kCsplit * kWaves;  // 32 waves stripe the tiles

DEVFN float softplus_f(float x) { return x > 20.0f ? x : log1pf(__expf(x)); }

// A-matrix (16x32 f16) lane/K order chosen here (contiguous, same as B; not the ISA table order):
// lane half 0: K in [0,16); half 1: K in [16,32); 2 halfs per VGPR.
DEVFN int kpatA(int r, int hf) { return 16*hf + 2*r; }

// ---------------------------------------------------------------------------
// Kernel 1: encoder + reparameterized z + closed-form KL. One block per sample.
// h[e] = sum_c relu( center.W_enc[e,:128] + ctx_c.W_enc[e,128:] + b_enc[e] )
// ---------------------------------------------------------------------------
__global__ void __launch_bounds__(256)
enc_kernel(const int* __restrict__ center_id, const int* __restrict__ context_ids,
           const float* __restrict__ epsilon, const float* __restrict__ embeddings,
           const float* __restrict__ prior_locs, const float* __restrict__ prior_scales,
           const float* __restrict__ W_enc, const float* __restrict__ b_enc,
           const float* __restrict__ W_loc, const float* __restrict__ b_loc,
           const float* __restrict__ W_scale, const float* __restrict__ b_scale,
           float* __restrict__ z_out, float* __restrict__ kl_out)
{
  __shared__ float sC[kD];
  __shared__ float sX[kC][kD];
  __shared__ float sH[kE];
  __shared__ float sLoc[kD];
  __shared__ float sSp[kD];
  __shared__ float sRed[kD];

  const int b   = blockIdx.x;
  const int tid = threadIdx.x;
  const int cid = center_id[b];

  if (tid < kD) sC[tid] = embeddings[(size_t)cid * kD + tid];
  for (int i = tid; i < kC * kD; i += 256) {
    const int c = i >> 7, d = i & 127;
    const int j = context_ids[b * kC + c];
    sX[c][d] = embeddings[(size_t)j * kD + d];
  }
  __syncthreads();

  // Each thread owns one encoder output channel e (W_enc rows are L2-resident).
  {
    const int e = tid;
    const float* wrow = W_enc + (size_t)e * kE;
    float a = b_enc[e];
    for (int d = 0; d < kD; ++d) a += sC[d] * wrow[d];
    float t[kC];
#pragma unroll
    for (int c = 0; c < kC; ++c) t[c] = 0.0f;
    for (int d = 0; d < kD; ++d) {
      const float w = wrow[kD + d];
#pragma unroll
      for (int c = 0; c < kC; ++c) t[c] += sX[c][d] * w;
    }
    float hs = 0.0f;
#pragma unroll
    for (int c = 0; c < kC; ++c) { const float v = a + t[c]; hs += (v > 0.0f) ? v : 0.0f; }
    sH[e] = hs;
  }
  __syncthreads();

  // loc (threads 0..127) and pre-softplus scale (threads 128..255)
  if (tid < kD) {
    const float* wl = W_loc + (size_t)tid * kE;
    float acc = b_loc[tid];
    for (int k = 0; k < kE; ++k) acc += sH[k] * wl[k];
    sLoc[tid] = acc;
  } else {
    const int i = tid - kD;
    const float* wsc = W_scale + (size_t)i * kE;
    float acc = b_scale[i];
    for (int k = 0; k < kE; ++k) acc += sH[k] * wsc[k];
    sSp[i] = acc;
  }
  __syncthreads();

  if (tid < kD) {
    const float loc   = sLoc[tid];
    const float scale = softplus_f(sSp[tid]);
    const float zz    = loc + scale * epsilon[tid];
    z_out[(size_t)b * kD + tid] = zz;
    const float ploc = prior_locs[(size_t)cid * kD + tid];
    const float psc  = softplus_f(prior_scales[(size_t)cid * kD + tid]);
    const float dl   = loc - ploc;
    sRed[tid] = __logf(psc / scale) + (scale * scale + dl * dl) / (2.0f * psc * psc) - 0.5f;
  }
  __syncthreads();
  for (int off = 64; off > 0; off >>= 1) {
    if (tid < off) sRed[tid] += sRed[tid + off];
    __syncthreads();
  }
  if (tid == 0) kl_out[b] = sRed[0];
}

// ---------------------------------------------------------------------------
// Kernel 2 (optional): one-time f32 -> f16 conversion of W_vocab and z.
// ---------------------------------------------------------------------------
__global__ void __launch_bounds__(256)
convert_f16_kernel(const float* __restrict__ z, const float* __restrict__ wv,
                   _Float16* __restrict__ zh, _Float16* __restrict__ wvh)
{
  const long total = (long)kV * kD;
  const long stride = (long)gridDim.x * 256;
  for (long i = (long)blockIdx.x * 256 + threadIdx.x; i < total; i += stride)
    wvh[i] = (_Float16)wv[i];
  for (int i = blockIdx.x * 256 + threadIdx.x; i < kB * kD; i += (int)stride)
    zh[i] = (_Float16)z[i];
}

// ---------------------------------------------------------------------------
// Kernel 3: logits = z @ W_vocab^T via v_wmma_f32_16x16x32_f16 with a fused
// online (flash-style) logsumexp.
//   grid = (B/16, kCsplit): block owns 16 batch rows and 1/4 of the vocab.
//   8 waves per block stripe that column range; B-fragments are double
//   buffered in registers (ping-pong) so next-tile loads overlap WMMA+VALU.
//   Block reduces to one (max,sumexp) per row per column-group -> workspace.
// ---------------------------------------------------------------------------
template <bool USE_F16>
__global__ void __launch_bounds__(256)
lse_kernel(const float* __restrict__ z32, const float* __restrict__ wv32,
           const uint32_t* __restrict__ zh, const uint32_t* __restrict__ wvh,
           const float* __restrict__ b_vocab,
           float* __restrict__ pm_out, float* __restrict__ ps_out)
{
  __shared__ float sM[16 * 128];
  __shared__ float sS[16 * 128];

  const int tid  = threadIdx.x;
  const int lane = tid & 31;
  const int wid  = tid >> 5;     // wave 0..7
  const int hf   = lane >> 4;    // lane half
  const int cls  = lane & 15;    // column residue class / A row select
  const int rb   = blockIdx.x * 16;
  const int g    = blockIdx.y;   // column group 0..3
  const int arow = rb + cls;

  union V16 { v16h v; uint32_t u[8]; };

  // A fragments: z rows, loaded once, reused for all tiles of this wave.
  V16 A[4];
#pragma unroll
  for (int q = 0; q < 4; ++q) {
#pragma unroll
    for (int r = 0; r < 8; ++r) {
      const int k = 32 * q + kpatA(r, hf);
      if constexpr (USE_F16) {
        A[q].u[r] = zh[arow * 64 + (k >> 1)];
      } else {
        const float2 p = ((const float2*)z32)[arow * 64 + (k >> 1)];
        A[q].v[2 * r]     = (_Float16)p.x;
        A[q].v[2 * r + 1] = (_Float16)p.y;
      }
    }
  }

  float m[8], s[8];
#pragma unroll
  for (int r = 0; r < 8; ++r) { m[r] = -3.0e38f; s[r] = 0.0f; }

  // B layout: lane = column, K = 32q + 16*half + [0,16): contiguous -> b128 loads.
  auto loadB = [&](int t, V16* Bf, float& bias) {
    const int col = t * 16 + cls;
    bias = b_vocab[col];
    if (t + kStride < kTiles) {   // wave-uniform long-range prefetch
      if constexpr (USE_F16) __builtin_prefetch(&wvh[(size_t)(col + 16 * kStride) * 64], 0, 1);
      else                   __builtin_prefetch(&wv32[(size_t)(col + 16 * kStride) * 128], 0, 1);
    }
#pragma unroll
    for (int q = 0; q < 4; ++q) {
      if constexpr (USE_F16) {
        const uint4* p4 = (const uint4*)wvh;
        const int base = col * 16 + 4 * q + 2 * hf;
        const uint4 lo = p4[base], hi = p4[base + 1];
        Bf[q].u[0] = lo.x; Bf[q].u[1] = lo.y; Bf[q].u[2] = lo.z; Bf[q].u[3] = lo.w;
        Bf[q].u[4] = hi.x; Bf[q].u[5] = hi.y; Bf[q].u[6] = hi.z; Bf[q].u[7] = hi.w;
      } else {
        const float4* p4 = (const float4*)wv32;
        const int base = col * 32 + 8 * q + 4 * hf;
        const float4 f0 = p4[base], f1 = p4[base + 1], f2 = p4[base + 2], f3 = p4[base + 3];
        Bf[q].v[0]=(_Float16)f0.x;  Bf[q].v[1]=(_Float16)f0.y;  Bf[q].v[2]=(_Float16)f0.z;  Bf[q].v[3]=(_Float16)f0.w;
        Bf[q].v[4]=(_Float16)f1.x;  Bf[q].v[5]=(_Float16)f1.y;  Bf[q].v[6]=(_Float16)f1.z;  Bf[q].v[7]=(_Float16)f1.w;
        Bf[q].v[8]=(_Float16)f2.x;  Bf[q].v[9]=(_Float16)f2.y;  Bf[q].v[10]=(_Float16)f2.z; Bf[q].v[11]=(_Float16)f2.w;
        Bf[q].v[12]=(_Float16)f3.x; Bf[q].v[13]=(_Float16)f3.y; Bf[q].v[14]=(_Float16)f3.z; Bf[q].v[15]=(_Float16)f3.w;
      }
    }
  };

  auto computeTile = [&](const V16* Bf, float bias) {
    v8f c = {};
#pragma unroll
    for (int q = 0; q < 4; ++q)
      c = __builtin_amdgcn_wmma_f32_16x16x32_f16(false, A[q].v, false, Bf[q].v,
                                                 (short)0, c, false, false);
#pragma unroll
    for (int r = 0; r < 8; ++r) {        // online softmax update
      const float x  = c[r] + bias;
      const float nm = fmaxf(m[r], x);
      s[r] = s[r] * __expf(m[r] - nm) + __expf(x - nm);
      m[r] = nm;
    }
  };

  // Software-pipelined ping-pong over this wave's tile stripe.
  // Stripe start: g*8 + wid in [0,32); stride 32. Always >= 1 tile (32 <= 3125).
  V16 B0[4], B1[4];
  float bias0, bias1;
  int t = g * kWaves + wid;
  loadB(t, B0, bias0);
  while (true) {
    const int t1 = t + kStride;
    if (t1 < kTiles) loadB(t1, B1, bias1);
    computeTile(B0, bias0);
    if (t1 >= kTiles) break;
    const int t2 = t1 + kStride;
    if (t2 < kTiles) loadB(t2, B0, bias0);
    computeTile(B1, bias1);
    if (t2 >= kTiles) break;
    t = t2;
  }

  // C/D layout: VGPR r holds row r (lanes 0-15) / row r+8 (lanes 16-31).
#pragma unroll
  for (int r = 0; r < 8; ++r) {
    const int row = r + 8 * hf;
    sM[row * 128 + wid * 16 + cls] = m[r];
    sS[row * 128 + wid * 16 + cls] = s[r];
  }
  __syncthreads();
  if (tid < 16) {                         // merge 8 waves x 16 classes, fixed order
    float M = -3.0e38f, S = 0.0f;
    for (int i = 0; i < 128; ++i) {
      const float mi = sM[tid * 128 + i], si = sS[tid * 128 + i];
      const float nm = fmaxf(M, mi);
      S = S * __expf(M - nm) + si * __expf(mi - nm);
      M = nm;
    }
    pm_out[(size_t)(rb + tid) * kCsplit + g] = M;
    ps_out[(size_t)(rb + tid) * kCsplit + g] = S;
  }
}

// ---------------------------------------------------------------------------
// Kernel 4: merge the kCsplit column-group (max,sumexp) partials per row.
// ---------------------------------------------------------------------------
__global__ void __launch_bounds__(256)
lse_merge_kernel(const float* __restrict__ pm, const float* __restrict__ ps,
                 float* __restrict__ lse)
{
  const int row = blockIdx.x * 256 + threadIdx.x;
  if (row < kB) {
    float M = -3.0e38f, S = 0.0f;
#pragma unroll
    for (int gidx = 0; gidx < kCsplit; ++gidx) {
      const float mi = pm[(size_t)row * kCsplit + gidx];
      const float si = ps[(size_t)row * kCsplit + gidx];
      const float nm = fmaxf(M, mi);
      S = S * __expf(M - nm) + si * __expf(mi - nm);
      M = nm;
    }
    lse[row] = M + __logf(S);
  }
}

// ---------------------------------------------------------------------------
// Kernel 5: per-sample gather of context logits (f32) + assemble -elbo.
// ---------------------------------------------------------------------------
__global__ void __launch_bounds__(256)
gather_kernel(const int* __restrict__ context_ids, const float* __restrict__ z,
              const float* __restrict__ W_vocab, const float* __restrict__ b_vocab,
              const float* __restrict__ lse, const float* __restrict__ kl,
              float* __restrict__ nege)
{
  __shared__ float sK[kC];
  const int b = blockIdx.x;
  const int tid = threadIdx.x;
  const int lane = tid & 31, wid = tid >> 5;
  for (int k = wid; k < kC; k += 8) {
    const int j = context_ids[b * kC + k];
    float p = 0.0f;
    for (int d = lane; d < kD; d += 32)
      p += z[(size_t)b * kD + d] * W_vocab[(size_t)j * kD + d];
    for (int off = 16; off > 0; off >>= 1)
      p += __shfl_xor(p, off, 32);
    if (lane == 0) sK[k] = p + b_vocab[j];
  }
  __syncthreads();
  if (tid == 0) {
    float gsum = 0.0f;
    for (int k = 0; k < kC; ++k) gsum += sK[k];       // fixed-order, deterministic
    const float lls = gsum - (float)kC * lse[b];
    nege[b] = kl[b] - lls;                            // -(lls - kl)
  }
}

// ---------------------------------------------------------------------------
// Kernel 6: deterministic mean over batch -> scalar output.
// ---------------------------------------------------------------------------
__global__ void __launch_bounds__(256)
reduce_kernel(const float* __restrict__ nege, float* __restrict__ out)
{
  __shared__ float sR[256];
  const int tid = threadIdx.x;
  float a = 0.0f;
  for (int b = tid; b < kB; b += 256) a += nege[b];
  sR[tid] = a;
  __syncthreads();
  for (int off = 128; off > 0; off >>= 1) {
    if (tid < off) sR[tid] += sR[tid + off];
    __syncthreads();
  }
  if (tid == 0) out[0] = sR[0] * (1.0f / (float)kB);
}

}  // namespace

extern "C" void kernel_launch(void* const* d_in, const int* in_sizes, int n_in,
                              void* d_out, int out_size, void* d_ws, size_t ws_size,
                              hipStream_t stream)
{
  const int*   center_id    = (const int*)  d_in[0];
  const int*   context_ids  = (const int*)  d_in[1];
  const float* epsilon      = (const float*)d_in[2];
  const float* embeddings   = (const float*)d_in[3];
  const float* prior_locs   = (const float*)d_in[4];
  const float* prior_scales = (const float*)d_in[5];
  const float* W_enc        = (const float*)d_in[6];
  const float* b_enc        = (const float*)d_in[7];
  const float* W_loc        = (const float*)d_in[8];
  const float* b_loc        = (const float*)d_in[9];
  const float* W_scale      = (const float*)d_in[10];
  const float* b_scale      = (const float*)d_in[11];
  const float* W_vocab      = (const float*)d_in[12];
  const float* b_vocab      = (const float*)d_in[13];
  (void)in_sizes; (void)n_in; (void)out_size;

  // Workspace carve-up (all buffers fully rewritten every call).
  char* p = (char*)d_ws;
  float* z    = (float*)p;  p += (size_t)kB * kD * sizeof(float);
  float* kl   = (float*)p;  p += (size_t)kB * sizeof(float);
  float* lse  = (float*)p;  p += (size_t)kB * sizeof(float);
  float* nege = (float*)p;  p += (size_t)kB * sizeof(float);
  float* pm   = (float*)p;  p += (size_t)kB * kCsplit * sizeof(float);
  float* ps   = (float*)p;  p += (size_t)kB * kCsplit * sizeof(float);
  const size_t base_need = (size_t)(p - (char*)d_ws);
  const size_t f16_need  = base_need + (size_t)kB * kD * 2 + (size_t)kV * kD * 2;
  const bool use_f16_ws  = (ws_size >= f16_need);
  _Float16* zh  = nullptr;
  _Float16* wvh = nullptr;
  if (use_f16_ws) {
    zh  = (_Float16*)p;  p += (size_t)kB * kD * 2;
    wvh = (_Float16*)p;
  }

  enc_kernel<<<dim3(kB), dim3(256), 0, stream>>>(
      center_id, context_ids, epsilon, embeddings, prior_locs, prior_scales,
      W_enc, b_enc, W_loc, b_loc, W_scale, b_scale, z, kl);

  if (use_f16_ws) {
    convert_f16_kernel<<<dim3(4096), dim3(256), 0, stream>>>(z, W_vocab, zh, wvh);
    lse_kernel<true><<<dim3(kB / 16, kCsplit), dim3(256), 0, stream>>>(
        z, W_vocab, (const uint32_t*)zh, (const uint32_t*)wvh, b_vocab, pm, ps);
  } else {
    lse_kernel<false><<<dim3(kB / 16, kCsplit), dim3(256), 0, stream>>>(
        z, W_vocab, nullptr, nullptr, b_vocab, pm, ps);
  }

  lse_merge_kernel<<<dim3((kB + 255) / 256), dim3(256), 0, stream>>>(pm, ps, lse);

  gather_kernel<<<dim3(kB), dim3(256), 0, stream>>>(
      context_ids, z, W_vocab, b_vocab, lse, kl, nege);

  reduce_kernel<<<dim3(1), dim3(256), 0, stream>>>(nege, (float*)d_out);
}
